// GCN_Trans_82033875353797
// MI455X (gfx1250) — compile-verified, matrix operand order checked by reading, not run on hardware
//
#include <hip/hip_runtime.h>
#include <hip/hip_bf16.h>
#include <math.h>
#include <stdint.h>

typedef __attribute__((ext_vector_type(16))) _Float16 v16h;
typedef __attribute__((ext_vector_type(8)))  _Float16 v8h;
typedef __attribute__((ext_vector_type(8)))  float    v8f;
typedef __attribute__((ext_vector_type(4)))  unsigned int u32x4;
typedef __attribute__((ext_vector_type(8)))  int i32x8;
typedef __attribute__((ext_vector_type(4)))  int i32x4;

#define DEV __device__ __forceinline__

// ---- problem constants (match reference) ----
#define Bn   32
#define An   64
#define DAn  78
#define Dn   128
#define Hn   8
#define Tn   512
#define Nn   (Bn * An)         // 2048
#define NE   (Bn * 256)        // 8192 edges
#define NEGV (-1.0e9f)
#define EPSV (1e-5f)

// ================= WMMA fragment helpers (CDNA5 16x16x32 f16) ================
// A/B fragment from row-major f16 [rows, K]: lane (l&15) holds one row; halves
// i<8 -> K=k0+i, halves i>=8 -> K=k0+8+i, k0 = 8*(lane>=16). Two b128 loads.
DEV v16h frag_h(const _Float16* __restrict__ p, int ld) {
  int lane = threadIdx.x & 31;
  const _Float16* r = p + (size_t)(lane & 15) * ld + ((lane >> 4) << 3);
  v8h lo = *(const v8h*)r;
  v8h hi = *(const v8h*)(r + 16);
  v16h f;
#pragma unroll
  for (int i = 0; i < 8; ++i) { f[i] = lo[i]; f[8 + i] = hi[i]; }
  return f;
}

// Logical K = 16 (high 16 K-slots zero) — one b128 load.
DEV v16h frag_h_k16(const _Float16* __restrict__ p, int ld) {
  int lane = threadIdx.x & 31;
  const _Float16* r = p + (size_t)(lane & 15) * ld + ((lane >> 4) << 3);
  v8h lo = *(const v8h*)r;
  v16h f;
#pragma unroll
  for (int i = 0; i < 8; ++i) { f[i] = lo[i]; f[8 + i] = (_Float16)0.0f; }
  return f;
}

// ===================== Tensor Data Mover: 2D f16 tile -> LDS =================
// Loads `rows` rows of `rowlen` f16 elements (row stride `ld` elements) from
// global into LDS at byte offset lds_off, packed contiguously (rowlen*2B/row).
DEV void tdm_load_2d_f16(const _Float16* gsrc, unsigned lds_off,
                         unsigned rows, unsigned rowlen, unsigned ld) {
  unsigned long long ga = (unsigned long long)(uintptr_t)gsrc;
  u32x4 g0;
  g0[0] = 1u;                                    // count=1 (valid), user mode
  g0[1] = lds_off;                               // lds_addr (bytes)
  g0[2] = (unsigned)(ga & 0xffffffffu);          // global_addr[31:0]
  g0[3] = (unsigned)((ga >> 32) & 0x01ffffffu) | (2u << 30);  // addr[56:32]|type=2
  i32x8 g1;
  g1[0] = (int)(1u << 16);                       // wg_mask=0, data_size=1 (2B)
  g1[1] = (int)((ld & 0xffffu) << 16);           // tensor_dim0[15:0]
  g1[2] = (int)((ld >> 16) | ((rows & 0xffffu) << 16));       // dim0 hi | dim1 lo
  g1[3] = (int)((rows >> 16) | ((rowlen & 0xffffu) << 16));   // dim1 hi | tile_dim0
  g1[4] = (int)(rows & 0xffffu);                 // tile_dim1 | tile_dim2=0
  g1[5] = (int)ld;                               // tensor_dim0_stride[31:0]
  g1[6] = 0;                                     // stride hi | dim1_stride lo
  g1[7] = 0;
  i32x4 z4 = {0, 0, 0, 0};
#if __clang_major__ >= 23
  i32x8 z8 = {0, 0, 0, 0, 0, 0, 0, 0};
  __builtin_amdgcn_tensor_load_to_lds(g0, g1, z4, z4, z8, 0);
#else
  __builtin_amdgcn_tensor_load_to_lds(g0, g1, z4, z4, 0);
#endif
}

// ======== WMMA GEMM, 2x2 tile blocked: C = act(A[M,K]h @ W[N,K]h^T + b) ======
__global__ void gemm_wmma_kernel(const _Float16* __restrict__ A,
                                 const _Float16* __restrict__ W,
                                 const float* __restrict__ bias,
                                 float* __restrict__ C32,
                                 _Float16* __restrict__ C16,
                                 int M, int N, int K, int relu) {
  int wave  = blockIdx.x * (blockDim.x >> 5) + (threadIdx.x >> 5);
  int nt    = N >> 5;
  int total = (M >> 5) * nt;
  if (wave >= total) return;                 // wave-uniform: EXEC stays full
  int m0 = (wave / nt) << 5;
  int n0 = (wave % nt) << 5;
  const _Float16* Ap0 = A + (size_t)m0 * K;
  const _Float16* Ap1 = A + (size_t)(m0 + 16) * K;
  const _Float16* Wp0 = W + (size_t)n0 * K;
  const _Float16* Wp1 = W + (size_t)(n0 + 16) * K;
  v8f c00 = {}, c01 = {}, c10 = {}, c11 = {};
  for (int k = 0; k < K; k += 32) {
    if (k + 32 < K) {                        // prefetch next K-slab
      __builtin_prefetch(Ap0 + k + 32, 0, 0);
      __builtin_prefetch(Wp0 + k + 32, 0, 0);
    }
    v16h a0 = frag_h(Ap0 + k, K);
    v16h a1 = frag_h(Ap1 + k, K);
    v16h b0 = frag_h(Wp0 + k, K);
    v16h b1 = frag_h(Wp1 + k, K);
    c00 = __builtin_amdgcn_wmma_f32_16x16x32_f16(false, a0, false, b0, (short)0, c00, false, false);
    c01 = __builtin_amdgcn_wmma_f32_16x16x32_f16(false, a0, false, b1, (short)0, c01, false, false);
    c10 = __builtin_amdgcn_wmma_f32_16x16x32_f16(false, a1, false, b0, (short)0, c10, false, false);
    c11 = __builtin_amdgcn_wmma_f32_16x16x32_f16(false, a1, false, b1, (short)0, c11, false, false);
  }
  int lane = threadIdx.x & 31;
  int col  = lane & 15;
  int rb   = (lane >> 4) << 3;
  auto emit = [&](v8f acc, int mi, int ni) {
    float bv = bias ? bias[n0 + ni + col] : 0.0f;
#pragma unroll
    for (int r = 0; r < 8; ++r) {
      float v = acc[r] + bv;
      if (relu) v = fmaxf(v, 0.0f);
      size_t idx = (size_t)(m0 + mi + rb + r) * N + n0 + ni + col;
      if (C32) C32[idx] = v;
      if (C16) C16[idx] = (_Float16)v;
    }
  };
  emit(c00, 0, 0); emit(c01, 0, 16); emit(c10, 16, 0); emit(c11, 16, 16);
}

// ===== fused flash attention; K/V head-slices staged to LDS via the TDM ======
// One block per (b,h): wave 0 issues two tensor_load_to_lds for the [Tk,16]
// f16 K and V slices, waits TENSORcnt, block syncs; 8 waves then split q-tiles.
__global__ void attn_fused_kernel(const _Float16* __restrict__ Q,
                                  const _Float16* __restrict__ Km,
                                  const _Float16* __restrict__ V,
                                  const int* __restrict__ ktok,  // mask: tok==0
                                  _Float16* __restrict__ O,
                                  int Tq, int Tk, int ldq, int ldk, int ldo,
                                  float sinv) {
  __shared__ __align__(16) _Float16 Ksh[Tn * 16];
  __shared__ __align__(16) _Float16 Vsh[Tn * 16];
  __shared__ float pbuf[8][256];
  int b = blockIdx.x >> 3;
  int h = blockIdx.x & 7;
  int wid  = threadIdx.x >> 5;
  int lane = threadIdx.x & 31;
  int col  = lane & 15;
  int rb   = (lane >> 4) << 3;

  if (wid == 0) {
    tdm_load_2d_f16(Km + (size_t)(b * Tk) * ldk + h * 16,
                    (unsigned)(uintptr_t)Ksh, Tk, 16, ldk);
    tdm_load_2d_f16(V + (size_t)(b * Tk) * ldk + h * 16,
                    (unsigned)(uintptr_t)Vsh, Tk, 16, ldk);
    __builtin_amdgcn_s_wait_tensorcnt(0);
  }
  __syncthreads();

  float* pw = pbuf[wid];
  int nqt = Tq >> 4;
  for (int qt = wid; qt < nqt; qt += 8) {
    int q0 = qt << 4;
    v16h qa = frag_h_k16(Q + (size_t)(b * Tq + q0) * ldq + h * 16, ldq);
    float mrow[8], lrow[8];
    v8f oacc = {};
#pragma unroll
    for (int r = 0; r < 8; ++r) { mrow[r] = -3.0e38f; lrow[r] = 0.0f; }

    for (int k0 = 0; k0 < Tk; k0 += 16) {
      v16h kb = frag_h_k16(Ksh + k0 * 16, 16);   // from LDS
      v8f s = {};
      s = __builtin_amdgcn_wmma_f32_16x16x32_f16(false, qa, false, kb,
                                                 (short)0, s, false, false);
      bool masked = ktok && (ktok[b * Tk + k0 + col] == 0);
#pragma unroll
      for (int r = 0; r < 8; ++r) {
        float v = s[r] * sinv;
        s[r] = masked ? NEGV : v;
      }
      // online softmax per row (16 lanes share a row -> xor butterflies)
#pragma unroll
      for (int r = 0; r < 8; ++r) {
        float mx = s[r];
        for (int m = 1; m < 16; m <<= 1) mx = fmaxf(mx, __shfl_xor(mx, m, 32));
        float mn = fmaxf(mrow[r], mx);
        float p  = __expf(s[r] - mn);
        float ps = p;
        for (int m = 1; m < 16; m <<= 1) ps += __shfl_xor(ps, m, 32);
        float corr = __expf(mrow[r] - mn);
        lrow[r] = lrow[r] * corr + ps;
        oacc[r] *= corr;
        mrow[r] = mn;
        s[r] = p;
      }
      // P: C-layout -> per-wave LDS slice -> A-fragment (DS is in-order per wave)
      __builtin_amdgcn_wave_barrier();
#pragma unroll
      for (int r = 0; r < 8; ++r) pw[(rb + r) * 16 + col] = s[r];
      __builtin_amdgcn_wave_barrier();
      v16h pa;
      {
        int row = lane & 15;
        int kk  = (lane >> 4) << 3;
#pragma unroll
        for (int i = 0; i < 8; ++i) pa[i] = (_Float16)pw[row * 16 + kk + i];
#pragma unroll
        for (int i = 0; i < 8; ++i) pa[8 + i] = (_Float16)0.0f;
      }
      __builtin_amdgcn_wave_barrier();
      // V B-fragment (row d = column d of LDS [16-wide] tile)
      v16h vb;
      {
        int d  = lane & 15;
        int kk = (lane >> 4) << 3;
#pragma unroll
        for (int i = 0; i < 8; ++i) vb[i] = Vsh[(k0 + kk + i) * 16 + d];
#pragma unroll
        for (int i = 0; i < 8; ++i) vb[8 + i] = (_Float16)0.0f;
      }
      oacc = __builtin_amdgcn_wmma_f32_16x16x32_f16(false, pa, false, vb,
                                                    (short)0, oacc, false, false);
    }
#pragma unroll
    for (int r = 0; r < 8; ++r)
      O[(size_t)(b * Tq + q0 + rb + r) * ldo + h * 16 + col] =
          (_Float16)(oacc[r] / lrow[r]);
  }
}

// ===================== residual + LayerNorm (D = 128) ========================
__global__ void add_ln_kernel(const float* __restrict__ X,
                              const float* __restrict__ Y,
                              const float* __restrict__ w,
                              const float* __restrict__ b,
                              float* __restrict__ Out32,
                              _Float16* __restrict__ Out16) {
  __shared__ float red[128];
  int row = blockIdx.x;
  int t   = threadIdx.x;
  float v = X[row * Dn + t] + Y[row * Dn + t];
  red[t] = v; __syncthreads();
  for (int s = 64; s > 0; s >>= 1) { if (t < s) red[t] += red[t + s]; __syncthreads(); }
  float mean = red[0] / Dn; __syncthreads();
  float d = v - mean;
  red[t] = d * d; __syncthreads();
  for (int s = 64; s > 0; s >>= 1) { if (t < s) red[t] += red[t + s]; __syncthreads(); }
  float var = red[0] / Dn;
  float o = d * rsqrtf(var + EPSV) * w[t] + b[t];
  Out32[row * Dn + t] = o;
  Out16[row * Dn + t] = (_Float16)o;
}

// ===================== small scalar GEMM (K not mult of 32) ==================
__global__ void small_gemm_kernel(const float* __restrict__ A,
                                  const float* __restrict__ W,
                                  const float* __restrict__ bias,
                                  float* __restrict__ C,
                                  _Float16* __restrict__ C16,
                                  int M, int N, int K, int relu) {
  int idx = blockIdx.x * blockDim.x + threadIdx.x;
  if (idx >= M * N) return;
  int m = idx / N, n = idx % N;
  float acc = bias ? bias[n] : 0.0f;
  const float* a = A + m * K;
  const float* w = W + n * K;
  for (int k = 0; k < K; ++k) acc += a[k] * w[k];
  if (relu) acc = fmaxf(acc, 0.0f);
  if (C)   C[idx] = acc;
  if (C16) C16[idx] = (_Float16)acc;
}

// ===================== f32 -> f16 conversion (weights) =======================
__global__ void cvt_h_kernel(const float* __restrict__ in,
                             _Float16* __restrict__ out, int n) {
  int i = blockIdx.x * blockDim.x + threadIdx.x;
  if (i < n) out[i] = (_Float16)in[i];
}

// ===================== GCN helpers ===========================================
__global__ void fill_kernel(float* p, int n, float v) {
  int i = blockIdx.x * blockDim.x + threadIdx.x;
  if (i < n) p[i] = v;
}

__global__ void deg_kernel(const int* __restrict__ dst, float* __restrict__ deg) {
  int e = blockIdx.x * blockDim.x + threadIdx.x;
  if (e < NE) atomicAdd(&deg[dst[e]], 1.0f);
}

__global__ void gcn_agg_kernel(const int* __restrict__ src,
                               const int* __restrict__ dst,
                               const float* __restrict__ deg,
                               const float* __restrict__ xw,
                               float* __restrict__ agg, int F) {
  int idx = blockIdx.x * blockDim.x + threadIdx.x;
  if (idx >= NE * F) return;
  int e = idx / F, f = idx % F;
  int s = src[e], d = dst[e];
  float coef = rsqrtf(deg[s]) * rsqrtf(deg[d]);
  atomicAdd(&agg[d * F + f], xw[s * F + f] * coef);
}

__global__ void gcn_combine_kernel(const float* __restrict__ xw,
                                   const float* __restrict__ agg,
                                   const float* __restrict__ deg,
                                   const float* __restrict__ b,
                                   float* __restrict__ out, int F) {
  int idx = blockIdx.x * blockDim.x + threadIdx.x;
  if (idx >= Nn * F) return;
  int v = idx / F, f = idx % F;
  out[idx] = fmaxf(agg[idx] + xw[idx] / deg[v] + b[f], 0.0f);
}

// ===================== misc ==================================================
__global__ void row_mean_kernel(const float* __restrict__ in,
                                float* __restrict__ out32,
                                _Float16* __restrict__ out16, int Tlen) {
  int idx = blockIdx.x * blockDim.x + threadIdx.x;
  if (idx >= Bn * Dn) return;
  int b = idx / Dn, d = idx % Dn;
  float s = 0.0f;
  for (int t = 0; t < Tlen; ++t) s += in[(b * Tlen + t) * Dn + d];
  float m = s / (float)Tlen;
  if (out32) out32[idx] = m;
  if (out16) out16[idx] = (_Float16)m;
}

__global__ void embed_pe_kernel(const int* __restrict__ seq,
                                const float* __restrict__ emb,
                                float* __restrict__ X32,
                                _Float16* __restrict__ X16) {
  int idx = blockIdx.x * blockDim.x + threadIdx.x;
  if (idx >= Bn * Tn * Dn) return;
  int d  = idx % Dn;
  int bt = idx / Dn;
  int t  = bt % Tn;
  int tok = seq[bt];
  float div = __expf((float)(d & ~1) * (-9.210340371976184f / (float)Dn));
  float ang = (float)t * div;
  float pe  = (d & 1) ? __cosf(ang) : __sinf(ang);
  float v = emb[tok * Dn + d] + pe;
  X32[idx] = v;
  X16[idx] = (_Float16)v;
}

// single-query distance attention (Tq == 1): one block per (b,h), f16 operands
__global__ void dist_attn_kernel(const _Float16* __restrict__ q,
                                 const _Float16* __restrict__ Kp,
                                 const _Float16* __restrict__ Vp,
                                 _Float16* __restrict__ o) {
  __shared__ float p[Tn];
  __shared__ float red[Tn];
  int bh = blockIdx.x;
  int b = bh >> 3, h = bh & 7;
  int t = threadIdx.x;
  const _Float16* qv = q + b * Dn + h * 16;
  const _Float16* kv = Kp + (size_t)(b * Tn + t) * Dn + h * 16;
  float s = 0.0f;
#pragma unroll
  for (int d = 0; d < 16; ++d) s += (float)qv[d] * (float)kv[d];
  s *= 0.25f;
  red[t] = s; __syncthreads();
  for (int st = 256; st > 0; st >>= 1) { if (t < st) red[t] = fmaxf(red[t], red[t + st]); __syncthreads(); }
  float mx = red[0]; __syncthreads();
  float pv = __expf(s - mx);
  red[t] = pv; __syncthreads();
  for (int st = 256; st > 0; st >>= 1) { if (t < st) red[t] += red[t + st]; __syncthreads(); }
  float sum = red[0];
  p[t] = pv / sum;
  __syncthreads();
  int g  = t >> 5;
  int ln = t & 31;
  float acc = 0.0f;
  for (int tt = ln; tt < Tn; tt += 32)
    acc += p[tt] * (float)Vp[(size_t)(b * Tn + tt) * Dn + h * 16 + g];
  for (int m = 1; m < 32; m <<= 1) acc += __shfl_xor(acc, m, 32);
  if (ln == 0) o[b * Dn + h * 16 + g] = (_Float16)acc;
}

__global__ void concat3_kernel(const float* a, const float* b, const float* c,
                               _Float16* out) {
  int idx = blockIdx.x * blockDim.x + threadIdx.x;
  if (idx >= Bn * 3 * Dn) return;
  int bi = idx / (3 * Dn), r = idx % (3 * Dn);
  float v = (r < Dn) ? a[bi * Dn + r]
          : (r < 2 * Dn) ? b[bi * Dn + r - Dn]
          : c[bi * Dn + r - 2 * Dn];
  out[idx] = (_Float16)v;
}

__global__ void final_out_kernel(const float* __restrict__ f2,
                                 const float* __restrict__ ow,
                                 const float* __restrict__ ob,
                                 float* __restrict__ out) {
  int b = threadIdx.x;
  if (b >= Bn) return;
  float s = ob[0];
  for (int d = 0; d < Dn; ++d) s += f2[b * Dn + d] * ow[d];
  out[b] = s;
}

// =============================================================================
extern "C" void kernel_launch(void* const* d_in, const int* in_sizes, int n_in,
                              void* d_out, int out_size, void* d_ws, size_t ws_size,
                              hipStream_t stream) {
  (void)in_sizes; (void)n_in; (void)out_size; (void)ws_size;
  const float* drug_x  = (const float*)d_in[0];
  const float* gcn1_w  = (const float*)d_in[2];
  const float* gcn1_b  = (const float*)d_in[3];
  const float* gcn2_w  = (const float*)d_in[4];
  const float* gcn2_b  = (const float*)d_in[5];
  const float* fcg1_w  = (const float*)d_in[6];
  const float* fcg1_b  = (const float*)d_in[7];
  const float* fcg2_w  = (const float*)d_in[8];
  const float* fcg2_b  = (const float*)d_in[9];
  const float* emb     = (const float*)d_in[10];
  const float* t_inw   = (const float*)d_in[11];
  const float* t_inb   = (const float*)d_in[12];
  const float* t_outw  = (const float*)d_in[13];
  const float* t_outb  = (const float*)d_in[14];
  const float* t_l1w   = (const float*)d_in[15];
  const float* t_l1b   = (const float*)d_in[16];
  const float* t_l2w   = (const float*)d_in[17];
  const float* t_l2b   = (const float*)d_in[18];
  const float* t_ln1w  = (const float*)d_in[19];
  const float* t_ln1b  = (const float*)d_in[20];
  const float* t_ln2w  = (const float*)d_in[21];
  const float* t_ln2b  = (const float*)d_in[22];
  const float* rnafc_w = (const float*)d_in[23];
  const float* rnafc_b = (const float*)d_in[24];
  const float* mhad_w  = (const float*)d_in[25];
  const float* mhad_b  = (const float*)d_in[26];
  const float* mhar_w  = (const float*)d_in[27];
  const float* mhar_b  = (const float*)d_in[28];
  const float* dist_w  = (const float*)d_in[29];
  const float* dist_b  = (const float*)d_in[30];
  const float* fc1_w   = (const float*)d_in[31];
  const float* fc1_b   = (const float*)d_in[32];
  const float* fc2_w   = (const float*)d_in[33];
  const float* fc2_b   = (const float*)d_in[34];
  const float* out_w   = (const float*)d_in[35];
  const float* out_b   = (const float*)d_in[36];
  const int* edge_idx  = (const int*)d_in[37];
  const int* rna_seq   = (const int*)d_in[39];
  const int* e_src = edge_idx;
  const int* e_dst = edge_idx + NE;

  float* ws = (float*)d_ws;
  size_t off = 0;
  auto allocf = [&](size_t n) { size_t a = (n + 3) & ~(size_t)3; float* p = ws + off; off += a; return p; };
  auto alloch = [&](size_t n) { size_t a = ((n + 1) / 2 + 3) & ~(size_t)3; _Float16* p = (_Float16*)(ws + off); off += a; return p; };

  // f32 scratch
  float* deg   = allocf(Nn);
  float* gA    = allocf((size_t)Nn * 156);
  float* gB    = allocf((size_t)Nn * 156);
  float* gH    = allocf((size_t)Nn * 156);
  float* da32  = allocf((size_t)Nn * Dn);
  float* SP32  = allocf((size_t)Nn * Dn);
  float* X32   = allocf((size_t)Bn * Tn * Dn);
  float* T2    = allocf((size_t)Bn * Tn * Dn);   // residual-path gemm outputs
  float* vecd  = allocf(Bn * Dn);
  float* vecr  = allocf(Bn * Dn);
  float* veco  = allocf(Bn * Dn);
  float* f2    = allocf(Bn * Dn);
  // f16 weight arena
  _Float16* fcg2_wh = alloch(Dn * Dn);
  _Float16* t_inwh  = alloch((size_t)2 * 3 * Dn * Dn);
  _Float16* t_outwh = alloch((size_t)2 * Dn * Dn);
  _Float16* t_l1wh  = alloch((size_t)2 * Dn * Dn);
  _Float16* t_l2wh  = alloch((size_t)2 * Dn * Dn);
  _Float16* rnafc_wh= alloch(Dn * Dn);
  _Float16* mhad_wh = alloch((size_t)4 * Dn * Dn);
  _Float16* mhar_wh = alloch((size_t)4 * Dn * Dn);
  _Float16* dist_wh = alloch((size_t)4 * Dn * Dn);
  _Float16* fc1_wh  = alloch((size_t)Dn * 3 * Dn);
  _Float16* fc2_wh  = alloch(Dn * Dn);
  // f16 activations
  _Float16* Xh    = alloch((size_t)Bn * Tn * Dn);
  _Float16* QKVh  = alloch((size_t)Bn * Tn * 3 * Dn);
  _Float16* BigH1 = alloch((size_t)Bn * Tn * Dn);   // sa / cross-r q
  _Float16* BigH2 = alloch((size_t)Bn * Tn * Dn);   // cross-r attn out
  _Float16* T3h   = alloch((size_t)Bn * Tn * Dn);   // ffn mid
  _Float16* RNAoh = alloch((size_t)Bn * Tn * Dn);
  _Float16* KBh   = alloch((size_t)Bn * Tn * Dn);
  _Float16* VBh   = alloch((size_t)Bn * Tn * Dn);
  _Float16* dah   = alloch((size_t)Nn * Dn);
  _Float16* QSh   = alloch((size_t)Nn * Dn);
  _Float16* KSh   = alloch((size_t)Nn * Dn);
  _Float16* CDh   = alloch((size_t)Nn * Dn);
  _Float16* poolh = alloch(Bn * Dn);
  _Float16* dpoolh= alloch(Bn * Dn);
  _Float16* q32h  = alloch(Bn * Dn);
  _Float16* o32h  = alloch(Bn * Dn);
  _Float16* xcbh  = alloch(Bn * 3 * Dn);
  _Float16* f1h   = alloch(Bn * Dn);

  auto grid1 = [](int n) { return (n + 255) / 256; };
  auto gemm = [&](const _Float16* A, const _Float16* W, const float* bias,
                  float* C32, _Float16* C16, int M, int N, int K, int relu) {
    int waves  = (M >> 5) * (N >> 5);
    int blocks = (waves + 3) / 4;
    gemm_wmma_kernel<<<blocks, 128, 0, stream>>>(A, W, bias, C32, C16, M, N, K, relu);
  };
  auto cvt = [&](const float* s, _Float16* d, int n) {
    cvt_h_kernel<<<grid1(n), 256, 0, stream>>>(s, d, n);
  };

  // -------- weights to f16 (one pass) --------
  cvt(fcg2_w, fcg2_wh, Dn * Dn);
  cvt(t_inw,  t_inwh,  2 * 3 * Dn * Dn);
  cvt(t_outw, t_outwh, 2 * Dn * Dn);
  cvt(t_l1w,  t_l1wh,  2 * Dn * Dn);
  cvt(t_l2w,  t_l2wh,  2 * Dn * Dn);
  cvt(rnafc_w, rnafc_wh, Dn * Dn);
  cvt(mhad_w, mhad_wh, 4 * Dn * Dn);
  cvt(mhar_w, mhar_wh, 4 * Dn * Dn);
  cvt(dist_w, dist_wh, 4 * Dn * Dn);
  cvt(fc1_w,  fc1_wh,  Dn * 3 * Dn);
  cvt(fc2_w,  fc2_wh,  Dn * Dn);

  // -------- GCN branch --------
  fill_kernel<<<grid1(Nn), 256, 0, stream>>>(deg, Nn, 1.0f);
  deg_kernel<<<grid1(NE), 256, 0, stream>>>(e_dst, deg);
  small_gemm_kernel<<<grid1(Nn * DAn), 256, 0, stream>>>(drug_x, gcn1_w, nullptr, gA, nullptr, Nn, DAn, DAn, 0);
  fill_kernel<<<grid1(Nn * DAn), 256, 0, stream>>>(gB, Nn * DAn, 0.0f);
  gcn_agg_kernel<<<grid1(NE * DAn), 256, 0, stream>>>(e_src, e_dst, deg, gA, gB, DAn);
  gcn_combine_kernel<<<grid1(Nn * DAn), 256, 0, stream>>>(gA, gB, deg, gcn1_b, gH, DAn);
  small_gemm_kernel<<<grid1(Nn * 156), 256, 0, stream>>>(gH, gcn2_w, nullptr, gA, nullptr, Nn, 156, DAn, 0);
  fill_kernel<<<grid1(Nn * 156), 256, 0, stream>>>(gB, Nn * 156, 0.0f);
  gcn_agg_kernel<<<grid1(NE * 156), 256, 0, stream>>>(e_src, e_dst, deg, gA, gB, 156);
  gcn_combine_kernel<<<grid1(Nn * 156), 256, 0, stream>>>(gA, gB, deg, gcn2_b, gH, 156);
  small_gemm_kernel<<<grid1(Nn * Dn), 256, 0, stream>>>(gH, fcg1_w, fcg1_b, da32, dah, Nn, Dn, 156, 1);
  row_mean_kernel<<<grid1(Bn * Dn), 256, 0, stream>>>(da32, nullptr, poolh, An);
  gemm(poolh, fcg2_wh, fcg2_b, nullptr, dpoolh, Bn, Dn, Dn, 0);

  // -------- RNA transformer --------
  embed_pe_kernel<<<grid1(Bn * Tn * Dn), 256, 0, stream>>>(rna_seq, emb, X32, Xh);
  for (int l = 0; l < 2; ++l) {
    gemm(Xh, t_inwh + (size_t)l * 3 * Dn * Dn, t_inb + l * 3 * Dn, nullptr, QKVh, Bn * Tn, 3 * Dn, Dn, 0);
    attn_fused_kernel<<<Bn * Hn, 256, 0, stream>>>(QKVh, QKVh + Dn, QKVh + 2 * Dn, rna_seq,
                                                   BigH1, Tn, Tn, 3 * Dn, 3 * Dn, Dn, 0.25f);
    gemm(BigH1, t_outwh + (size_t)l * Dn * Dn, t_outb + l * Dn, T2, nullptr, Bn * Tn, Dn, Dn, 0);
    add_ln_kernel<<<Bn * Tn, 128, 0, stream>>>(X32, T2, t_ln1w + l * Dn, t_ln1b + l * Dn, X32, Xh);
    gemm(Xh, t_l1wh + (size_t)l * Dn * Dn, t_l1b + l * Dn, nullptr, T3h, Bn * Tn, Dn, Dn, 1);
    gemm(T3h, t_l2wh + (size_t)l * Dn * Dn, t_l2b + l * Dn, T2, nullptr, Bn * Tn, Dn, Dn, 0);
    add_ln_kernel<<<Bn * Tn, 128, 0, stream>>>(X32, T2, t_ln2w + l * Dn, t_ln2b + l * Dn, X32, Xh);
  }
  gemm(Xh, rnafc_wh, rnafc_b, nullptr, RNAoh, Bn * Tn, Dn, Dn, 0);

  // -------- cross-attn: drug queries RNA --------
  gemm(dah,   mhad_wh + 0 * Dn * Dn, mhad_b + 0 * Dn, nullptr, QSh, Nn,      Dn, Dn, 0);
  gemm(RNAoh, mhad_wh + 1 * Dn * Dn, mhad_b + 1 * Dn, nullptr, KBh, Bn * Tn, Dn, Dn, 0);
  gemm(RNAoh, mhad_wh + 2 * Dn * Dn, mhad_b + 2 * Dn, nullptr, VBh, Bn * Tn, Dn, Dn, 0);
  attn_fused_kernel<<<Bn * Hn, 256, 0, stream>>>(QSh, KBh, VBh, rna_seq, CDh,
                                                 An, Tn, Dn, Dn, Dn, 0.25f);
  gemm(CDh, mhad_wh + 3 * Dn * Dn, mhad_b + 3 * Dn, SP32, nullptr, Nn, Dn, Dn, 0);
  row_mean_kernel<<<grid1(Bn * Dn), 256, 0, stream>>>(SP32, vecd, nullptr, An);

  // -------- cross-attn: RNA queries drug --------
  gemm(RNAoh, mhar_wh + 0 * Dn * Dn, mhar_b + 0 * Dn, nullptr, BigH1, Bn * Tn, Dn, Dn, 0);
  gemm(dah,   mhar_wh + 1 * Dn * Dn, mhar_b + 1 * Dn, nullptr, QSh, Nn, Dn, Dn, 0);
  gemm(dah,   mhar_wh + 2 * Dn * Dn, mhar_b + 2 * Dn, nullptr, KSh, Nn, Dn, Dn, 0);
  attn_fused_kernel<<<Bn * Hn, 256, 0, stream>>>(BigH1, QSh, KSh, nullptr, BigH2,
                                                 Tn, An, Dn, Dn, Dn, 0.25f);
  gemm(BigH2, mhar_wh + 3 * Dn * Dn, mhar_b + 3 * Dn, T2, nullptr, Bn * Tn, Dn, Dn, 0);
  row_mean_kernel<<<grid1(Bn * Dn), 256, 0, stream>>>(T2, vecr, nullptr, Tn);

  // -------- distance-aware attention (single query) --------
  gemm(dpoolh, dist_wh + 0 * Dn * Dn, dist_b + 0 * Dn, nullptr, q32h, Bn, Dn, Dn, 0);
  gemm(RNAoh,  dist_wh + 1 * Dn * Dn, dist_b + 1 * Dn, nullptr, KBh, Bn * Tn, Dn, Dn, 0);
  gemm(RNAoh,  dist_wh + 2 * Dn * Dn, dist_b + 2 * Dn, nullptr, VBh, Bn * Tn, Dn, Dn, 0);
  dist_attn_kernel<<<Bn * Hn, Tn, 0, stream>>>(q32h, KBh, VBh, o32h);
  gemm(o32h, dist_wh + 3 * Dn * Dn, dist_b + 3 * Dn, veco, nullptr, Bn, Dn, Dn, 0);

  // -------- head --------
  concat3_kernel<<<grid1(Bn * 3 * Dn), 256, 0, stream>>>(vecd, vecr, veco, xcbh);
  gemm(xcbh, fc1_wh, fc1_b, nullptr, f1h, Bn, Dn, 3 * Dn, 1);
  gemm(f1h,  fc2_wh, fc2_b, f2, nullptr, Bn, Dn, Dn, 1);
  final_out_kernel<<<1, Bn, 0, stream>>>(f2, out_w, out_b, (float*)d_out);
}
